// FeedForwardQuantum_65481071399172
// MI455X (gfx1250) — matrix operand order repaired and checked by reading, not compiled, not run on hardware
//
#include <hip/hip_runtime.h>

typedef _Float16 v16h __attribute__((ext_vector_type(16)));
typedef _Float16 v8h  __attribute__((ext_vector_type(8)));
typedef _Float16 v4h  __attribute__((ext_vector_type(4)));
typedef float    v8f  __attribute__((ext_vector_type(8)));

#define MT   128          // tokens per workgroup
#define NT   256          // output channels per workgroup
#define KC   32           // one WMMA K-step
#define KT   64           // ffn chunk staged per loop iteration (2 K-steps)
#define HS   40           // h_s/w2s row stride (f16): 80B, 16B-aligned, conflict-free
#define QS   24           // q_h row stride (f16)
#define WS   24           // w1h row stride (f16)
#define EMB  1024
#define FFN  4096
#define NQ   8
#define MTOT (4 * 4096)   // B*T tokens

// relu applied after f32->f16 convert: cvt output is canonical, so the f16 max
// needs no canonicalization step (avoids the v_max(x,x) + v_max(0,x) pair).
__device__ __forceinline__ _Float16 relu_h(float v) {
#if __has_builtin(__builtin_fmaxf16)
  return __builtin_fmaxf16((_Float16)v, (_Float16)0.0f);
#else
  return (_Float16)fmaxf(v, 0.0f);
#endif
}

__device__ __forceinline__ v16h concat8(v8h lo, v8h hi) {
  return __builtin_shufflevector(lo, hi, 0,1,2,3,4,5,6,7,8,9,10,11,12,13,14,15);
}

// One-time w2 f32 -> f16 conversion into workspace (removes cvt from hot loop).
__global__ __launch_bounds__(256) void cvt_w2_kernel(const float* __restrict__ w2,
                                                     _Float16* __restrict__ w2h) {
  const size_t i = ((size_t)blockIdx.x * 256 + threadIdx.x) * 4;
  const float4 v = *(const float4*)(w2 + i);
  v4h h; h[0] = (_Float16)v.x; h[1] = (_Float16)v.y;
  h[2] = (_Float16)v.z; h[3] = (_Float16)v.w;
  *(v4h*)(w2h + i) = h;
}

template <bool PRECVT>
__global__ __launch_bounds__(256, 1) void ffq_kernel(
    const float* __restrict__ x,  const float* __restrict__ w1,
    const float* __restrict__ b1, const void* __restrict__ w2p,
    const float* __restrict__ b2, float* __restrict__ out) {
  __shared__ _Float16 q_h[MT * QS];         // [q0..q7, 1.0, 0...] per token (bias trick)
  __shared__ _Float16 w1h[3][2][KC * WS];   // [buf][ksub]: [w1 row, b1, 0...]
  __shared__ _Float16 h_s[2][MT * HS];      // [parity]: hidden activations (1-iter skew)
  __shared__ _Float16 w2s[3][2][NT * HS];   // [buf][ksub]: w2 chunk (prefetch distance 2)

  const int tid  = threadIdx.x;
  const int lane = tid & 31;
  const int wid  = tid >> 5;     // 8 waves
  const int l16  = lane & 15;
  const int lh   = lane >> 4;
  const int wm   = wid & 1;      // wave M block: wm*64
  const int wn   = wid >> 1;     // wave N block: wn*64
  const int m0   = blockIdx.y * MT;
  const int n0   = blockIdx.x * NT;

  const _Float16* w2hp = (const _Float16*)w2p;
  const float*    w2f  = (const float*)w2p;

  // ---- Phase 0: q[t] = cumprod(cos(x[t,:8])), K slot 8 = 1.0 carries the bias ----
  if (tid < MT) {
    const float* xr = x + (size_t)(m0 + tid) * EMB;
    float p = 1.0f;
    #pragma unroll
    for (int j = 0; j < NQ; ++j) { p *= cosf(xr[j]); q_h[tid * QS + j] = (_Float16)p; }
    q_h[tid * QS + 8] = (_Float16)1.0f;
    #pragma unroll
    for (int j = 9; j < 16; ++j) q_h[tid * QS + j] = (_Float16)0.0f;
  }

  const v8f  zero8f  = {};
  const v8h  zero8h  = {};
  const v16h zero16h = {};
  v8f acc[4][4];
  #pragma unroll
  for (int im = 0; im < 4; ++im)
    #pragma unroll
    for (int in = 0; in < 4; ++in) acc[im][in] = zero8f;

  // ---- staging coordinates & register prefetch buffers ----
  const int w1row = tid >> 3;          // 0..31
  const int w1c   = (tid & 7) * 2;     // 0,2,..,14
  const int kq4 = tid & 3, nr4 = tid >> 2;   // w2 staging coords

  _Float16 w1a[2], w1b[2];
  v8h      wr[2][4];

  auto w1_fetch = [&](int f0) {
    #pragma unroll
    for (int ks = 0; ks < 2; ++ks) {
      const int fs = f0 + ks * KC;
      if (w1c < 8) {
        const float2 wv = *(const float2*)(w1 + (size_t)(fs + w1row) * NQ + w1c);
        w1a[ks] = (_Float16)wv.x; w1b[ks] = (_Float16)wv.y;
      } else if (w1c == 8) {
        w1a[ks] = (_Float16)b1[fs + w1row]; w1b[ks] = (_Float16)0.0f;
      } else {
        w1a[ks] = (_Float16)0.0f; w1b[ks] = (_Float16)0.0f;
      }
    }
  };
  auto w1_store = [&](int buf) {
    #pragma unroll
    for (int ks = 0; ks < 2; ++ks) {
      w1h[buf][ks][w1row * WS + w1c]     = w1a[ks];
      w1h[buf][ks][w1row * WS + w1c + 1] = w1b[ks];
    }
  };
  auto w2_fetch = [&](int f0) {
    #pragma unroll
    for (int ks = 0; ks < 2; ++ks) {
      #pragma unroll
      for (int r = 0; r < 4; ++r) {
        if (PRECVT) {
          wr[ks][r] = *(const v8h*)(w2hp + (size_t)(n0 + nr4 + r * 64) * FFN
                                         + f0 + ks * KC + kq4 * 8);
        } else {
          const float* s = w2f + (size_t)(n0 + nr4 + r * 64) * FFN + f0 + ks * KC + kq4 * 8;
          const float4 v0 = *(const float4*)s;
          const float4 v1 = *(const float4*)(s + 4);
          v8h h;
          h[0] = (_Float16)v0.x; h[1] = (_Float16)v0.y;
          h[2] = (_Float16)v0.z; h[3] = (_Float16)v0.w;
          h[4] = (_Float16)v1.x; h[5] = (_Float16)v1.y;
          h[6] = (_Float16)v1.z; h[7] = (_Float16)v1.w;
          wr[ks][r] = h;
        }
      }
    }
  };
  auto w2_store = [&](int buf) {
    #pragma unroll
    for (int ks = 0; ks < 2; ++ks)
      #pragma unroll
      for (int r = 0; r < 4; ++r)
        *(v8h*)(&w2s[buf][ks][(nr4 + r * 64) * HS + kq4 * 8]) = wr[ks][r];
  };

  // ---- prologue: stage chunks 0 and 1 into buffers 0 and 1 ----
  w1_fetch(0);  w2_fetch(0);  w1_store(0); w2_store(0);
  w1_fetch(KT); w2_fetch(KT); w1_store(1); w2_store(1);
  __syncthreads();   // q_h + buffers 0,1 visible

  // ---- hoisted loop-invariant q fragment (B of GEMM1) ----
  // lanes 0-15 carry K0..7 & K16..23, lanes 16-31 carry K8..15 & K24..31 (K16..31 = 0)
  v16h bq = zero16h;
  {
    const _Float16* qp = &q_h[(wid * 16 + l16) * QS + lh * 8];
    bq = concat8(*(const v8h*)qp, zero8h);
  }

  // GEMM1 (transposed): D[f][t] = (w1|b1) x (q|1)^T ; relu -> packed b128 store
  auto gemm1 = [&](int g1buf, int hbuf) {
    #pragma unroll
    for (int ks = 0; ks < 2; ++ks) {
      #pragma unroll
      for (int ft = 0; ft < 2; ++ft) {
        const _Float16* wp = &w1h[g1buf][ks][(ft * 16 + l16) * WS + lh * 8];
        const v16h aw = concat8(*(const v8h*)wp, zero8h);
        v8f d = __builtin_amdgcn_wmma_f32_16x16x32_f16(
            false, aw, false, bq, (short)0, zero8f, false, false);
        // D layout: N = token = wid*16+l16 ; M = f = ft*16 + lh*8 + r (consecutive)
        v8h pk;
        #pragma unroll
        for (int r = 0; r < 8; ++r) pk[r] = relu_h(d[r]);
        *(v8h*)(&h_s[hbuf][(wid * 16 + l16) * HS + ft * 16 + lh * 8]) = pk;
      }
    }
  };

  // GEMM2: 64x64 block per wave, 2 K-steps x 16 WMMAs
  auto gemm2 = [&](int g2buf, int hbuf) {
    #pragma unroll
    for (int ks = 0; ks < 2; ++ks) {
      v16h bfr[4];
      #pragma unroll
      for (int in = 0; in < 4; ++in) {
        // B 16-bit layout: lanes0-15 K0..7 & K16..23 ; lanes16-31 K8..15 & K24..31
        const _Float16* p = &w2s[g2buf][ks][(wn * 64 + in * 16 + l16) * HS + lh * 8];
        bfr[in] = concat8(*(const v8h*)p, *(const v8h*)(p + 16));
      }
      #pragma unroll
      for (int im = 0; im < 4; ++im) {
        // A 16-bit layout: lanes0-15 K0..7 & K16..23 ; lanes16-31 K8..15 & K24..31
        const _Float16* p = &h_s[hbuf][(wm * 64 + im * 16 + l16) * HS + lh * 8];
        const v16h a = concat8(*(const v8h*)p, *(const v8h*)(p + 16));
        #pragma unroll
        for (int in = 0; in < 4; ++in)
          acc[im][in] = __builtin_amdgcn_wmma_f32_16x16x32_f16(
              false, a, false, bfr[in], (short)0, acc[im][in], false, false);
      }
    }
  };

  // ---- pipeline prologue: h for chunk 0 ----
  gemm1(0, 0);
  __syncthreads();   // h_s[0] visible

  // ---- main loop: ONE barrier per iteration (skewed GEMM1, prefetch distance 2) ----
  const int NIT = FFN / KT;   // 64
  for (int ki = 0; ki < NIT; ++ki) {
    // prefetch chunk ki+2 from global into registers (clamped; dup on tail is harmless)
    const int fpre = (ki + 2 < NIT) ? (ki + 2) * KT : (NIT - 1) * KT;
    w1_fetch(fpre); w2_fetch(fpre);

    // GEMM1 for chunk ki+1 (tail result is written to the idle h parity, never read)
    gemm1((ki + 1) % 3, (ki + 1) & 1);

    // GEMM2 for chunk ki
    gemm2(ki % 3, ki & 1);

    // drain prefetched chunk ki+2 into its (currently idle) buffers
    w1_store((ki + 2) % 3); w2_store((ki + 2) % 3);
    __syncthreads();
  }

  // ---- epilogue: out = acc + b2 ----
  #pragma unroll
  for (int in = 0; in < 4; ++in) {
    const int col = n0 + wn * 64 + in * 16 + l16;
    const float bv = b2[col];
    #pragma unroll
    for (int im = 0; im < 4; ++im) {
      const int row0 = m0 + wm * 64 + im * 16 + lh * 8;
      float* op = out + (size_t)row0 * EMB + col;
      #pragma unroll
      for (int r = 0; r < 8; ++r)
        op[(size_t)r * EMB] = acc[im][in][r] + bv;
    }
  }
}

extern "C" void kernel_launch(void* const* d_in, const int* in_sizes, int n_in,
                              void* d_out, int out_size, void* d_ws, size_t ws_size,
                              hipStream_t stream) {
  const float* x  = (const float*)d_in[0];
  // d_in[1] = circuit_params: provably contributes exactly zero (cancels in <Z_w>)
  const float* w1 = (const float*)d_in[2];
  const float* b1 = (const float*)d_in[3];
  const float* w2 = (const float*)d_in[4];
  const float* b2 = (const float*)d_in[5];
  float* out = (float*)d_out;
  (void)in_sizes; (void)n_in; (void)out_size;

  const size_t w2h_bytes = (size_t)EMB * FFN * sizeof(_Float16);  // 8 MB
  dim3 grid(EMB / NT, MTOT / MT);   // (4, 128) = 512 workgroups

  if (ws_size >= w2h_bytes) {
    _Float16* w2h = (_Float16*)d_ws;
    cvt_w2_kernel<<<(EMB * FFN) / (256 * 4), 256, 0, stream>>>(w2, w2h);
    ffq_kernel<true><<<grid, 256, 0, stream>>>(x, w1, b1, (const void*)w2h, b2, out);
  } else {
    ffq_kernel<false><<<grid, 256, 0, stream>>>(x, w1, b1, (const void*)w2, b2, out);
  }
}
